// QFNN_83906481095375
// MI455X (gfx1250) — matrix operand order repaired and checked by reading, not compiled, not run on hardware
//
#include <hip/hip_runtime.h>

typedef __attribute__((ext_vector_type(16))) __bf16 v16bf;
typedef __attribute__((ext_vector_type(8)))  __bf16 v8bf;
typedef __attribute__((ext_vector_type(8)))  float  v8f;
typedef __attribute__((ext_vector_type(8)))  unsigned short v8us;

// ---------- helpers ----------
__device__ __forceinline__ unsigned short f2us(float f) {
    unsigned u = __float_as_uint(f);
    u += 0x7FFFu + ((u >> 16) & 1u);           // RNE round to bf16
    return (unsigned short)(u >> 16);
}

// A-fragment: row-major row pointer (already + k0). lane group g holds
// K = {8g..8g+7} U {16+8g..16+8g+7}  -> chunks [g] and [2+g] of 8 halves.
__device__ __forceinline__ v16bf loadA(const unsigned short* base, int g) {
    const v8us* p = (const v8us*)base;
    v8bf lo = __builtin_bit_cast(v8bf, p[g]);
    v8bf hi = __builtin_bit_cast(v8bf, p[2 + g]);
    return __builtin_shufflevector(lo, hi, 0,1,2,3,4,5,6,7,8,9,10,11,12,13,14,15);
}
// B-fragment: B stored [N][K] row-major (or any 32-contiguous-K chunk), ptr + k0.
// lane group g holds K = {8g..8g+7} U {16+8g..16+8g+7} -> chunks [g] and [2+g].
__device__ __forceinline__ v16bf loadB(const unsigned short* base, int g) {
    const v8us* p = (const v8us*)base;
    v8bf lo = __builtin_bit_cast(v8bf, p[g]);
    v8bf hi = __builtin_bit_cast(v8bf, p[2 + g]);
    return __builtin_shufflevector(lo, hi, 0,1,2,3,4,5,6,7,8,9,10,11,12,13,14,15);
}

// ---------- kernel 0: pack weights to bf16 (and rearrange conv2 w) ----------
__global__ void pack_weights(const float* __restrict__ c2w, const float* __restrict__ fc1w,
                             const float* __restrict__ fc2w, const float* __restrict__ fc3w,
                             unsigned short* __restrict__ w2t, unsigned short* __restrict__ w1b,
                             unsigned short* __restrict__ w2fb, unsigned short* __restrict__ w3b) {
    int i = blockIdx.x * 256 + threadIdx.x;
    if (i < 9 * 64 * 32) {               // w2t[tap][oc][ic] <- c2w[oc][ic][ky][kx]
        int tap = i / (64 * 32);
        int rem = i - tap * 64 * 32;
        int oc = rem >> 5, ic = rem & 31;
        w2t[i] = f2us(c2w[(oc * 32 + ic) * 9 + tap]);
    }
    if (i < 128 * 3136) w1b[i]  = f2us(fc1w[i]);
    if (i < 64 * 128)   w2fb[i] = f2us(fc2w[i]);
    if (i < 16 * 64)    w3b[i]  = f2us(fc3w[i]);
}

// ---------- kernel 1: conv1 (1->32, 3x3 SAME) + relu + 2x2 maxpool ----------
// Output: zero-padded NHWC bf16 tile p1pad[B][16][18][32]:
//   row 0 = y=-1 halo, rows 1..14 = y=0..13, row 15 = y=14 halo
//   col 0 = x=-1 halo, cols 1..14 = x=0..13, cols 15..17 = x=14..16 halo
//   innermost = 32 input channels (contiguous for conv2 B fragments).
__global__ void conv1_relu_pool(const float* __restrict__ x, const float* __restrict__ w,
                                const float* __restrict__ b, unsigned short* __restrict__ p1pad) {
    __shared__ float sx[784];
    __shared__ float sw[288];
    __shared__ float sb[32];
    int bi = blockIdx.x;                          // batch index
    for (int i = threadIdx.x; i < 784; i += 256) sx[i] = x[bi * 784 + i];
    for (int i = threadIdx.x; i < 288; i += 256) sw[i] = w[i];
    if (threadIdx.x < 32) sb[threadIdx.x] = b[threadIdx.x];
    __syncthreads();
    for (int oc = 0; oc < 32; ++oc) {
        const float* wk = &sw[oc * 9];
        for (int t = threadIdx.x; t < 196; t += 256) {
            int my = t / 14, mx = t - my * 14;
            float mv = -3.4e38f;
            #pragma unroll
            for (int dy = 0; dy < 2; ++dy)
                #pragma unroll
                for (int dx = 0; dx < 2; ++dx) {
                    int yy = 2 * my + dy, xx = 2 * mx + dx;
                    float s = 0.f;
                    #pragma unroll
                    for (int ky = 0; ky < 3; ++ky)
                        #pragma unroll
                        for (int kx = 0; kx < 3; ++kx) {
                            int iy = yy + ky - 1, ix = xx + kx - 1;
                            if (iy >= 0 && iy < 28 && ix >= 0 && ix < 28)
                                s += sx[iy * 28 + ix] * wk[ky * 3 + kx];
                        }
                    mv = fmaxf(mv, s);
                }
            float val = fmaxf(mv + sb[oc], 0.f);
            p1pad[(((size_t)bi * 16 + my + 1) * 18 + (mx + 1)) * 32 + oc] = f2us(val);
        }
    }
}

// ---------- kernel 2: conv2 (32->64) implicit-GEMM WMMA + relu + pool -> bf16 ----------
// grid = B*7 workgroups, 256 threads = 8 waves. Wave (mt, ns): mt = oc tile (4),
// ns = row within the 2-row pixel pair. N cols = x 0..15 (14 valid).
// B operand comes branch-free from the zero-padded NHWC tile.
__global__ void conv2_wmma_pool(const unsigned short* __restrict__ p1pad,
                                const unsigned short* __restrict__ w2t,
                                const float* __restrict__ c2b, unsigned short* __restrict__ p2b) {
    __shared__ float lsd[64][2][16];
    int b  = blockIdx.x / 7;
    int ty = blockIdx.x - b * 7;
    int wave = threadIdx.x >> 5, lane = threadIdx.x & 31;
    int mt = wave & 3, ns = wave >> 2;
    int r = lane & 15, g = lane >> 4;
    int y = 2 * ty + ns;                           // output row in 14x14 map
    const unsigned short* pb = p1pad + (size_t)b * 16 * 18 * 32;
    v8f acc;
    #pragma unroll
    for (int i = 0; i < 8; ++i) acc[i] = 0.f;

    #pragma unroll
    for (int tap = 0; tap < 9; ++tap) {
        int ky = tap / 3, kx = tap - ky * 3;
        // padded coords: iy_p = y+ky (0..15), ix_p = r+kx (0..17)
        const unsigned short* bbase = pb + ((size_t)(y + ky) * 18 + (r + kx)) * 32;
        const unsigned short* abase = w2t + ((size_t)(tap * 64) + mt * 16 + r) * 32;
        v16bf a = loadA(abase, g);
        v16bf bf = loadB(bbase, g);
        acc = __builtin_amdgcn_wmma_f32_16x16x32_bf16(false, a, false, bf,
                                                      (short)0, acc, false, false);
    }
    #pragma unroll
    for (int v = 0; v < 8; ++v) {
        int oc = mt * 16 + v + 8 * g;
        if (r < 14) lsd[oc][ns][r] = fmaxf(acc[v] + c2b[oc], 0.f);
    }
    __syncthreads();
    for (int idx = threadIdx.x; idx < 64 * 7; idx += 256) {
        int oc = idx / 7, px = idx - oc * 7;
        float m = fmaxf(fmaxf(lsd[oc][0][2 * px], lsd[oc][0][2 * px + 1]),
                        fmaxf(lsd[oc][1][2 * px], lsd[oc][1][2 * px + 1]));
        p2b[(size_t)b * 3136 + oc * 49 + ty * 7 + px] = f2us(m);
    }
}

// ---------- kernel 3: generic out = act(A @ B^T + bias), bf16 in, bf16/f32 out ----------
// A: [M][K] bf16, B: [N][K] bf16, K % 32 == 0, M,N % 16 == 0. One 16x16 tile per wave.
__global__ void gemm_bias_act(const unsigned short* __restrict__ A, const unsigned short* __restrict__ Bm,
                              const float* __restrict__ bias,
                              float* __restrict__ outF, unsigned short* __restrict__ outB,
                              int M, int N, int K, int doRelu) {
    int wave = threadIdx.x >> 5, lane = threadIdx.x & 31;
    int tile = blockIdx.x * (blockDim.x >> 5) + wave;
    int nTilesN = N >> 4;
    int nTiles = (M >> 4) * nTilesN;
    if (tile >= nTiles) return;
    int m0 = (tile / nTilesN) << 4;
    int n0 = (tile % nTilesN) << 4;
    int r = lane & 15, g = lane >> 4;
    const unsigned short* aRow = A  + (size_t)(m0 + r) * K;
    const unsigned short* bRow = Bm + (size_t)(n0 + r) * K;
    v8f acc;
    #pragma unroll
    for (int i = 0; i < 8; ++i) acc[i] = 0.f;
    #pragma unroll 2
    for (int k0 = 0; k0 < K; k0 += 32) {
        if (k0 + 128 < K) {
            __builtin_prefetch(aRow + k0 + 128, 0, 1);
            __builtin_prefetch(bRow + k0 + 128, 0, 1);
        }
        v16bf a = loadA(aRow + k0, g);
        v16bf b = loadB(bRow + k0, g);
        acc = __builtin_amdgcn_wmma_f32_16x16x32_bf16(false, a, false, b,
                                                      (short)0, acc, false, false);
    }
    float bv = bias[n0 + r];
    #pragma unroll
    for (int v = 0; v < 8; ++v) {
        int row = m0 + v + 8 * g;
        float val = acc[v] + bv;
        if (doRelu) val = fmaxf(val, 0.f);
        if (outB) outB[(size_t)row * N + n0 + r] = f2us(val);
        else      outF[(size_t)row * N + n0 + r] = val;
    }
}

// ---------- kernel 4: quantum circuit + fc4 + log_softmax, 1 lane per sample ----------
__global__ void quantum_fc4(const float* __restrict__ feat, const float* __restrict__ qw,
                            const float* __restrict__ fc4w, const float* __restrict__ fc4b,
                            float* __restrict__ out) {
    int b = blockIdx.x * blockDim.x + threadIdx.x;   // 4 blocks * 256 = 1024 exactly
    float f[16];
    #pragma unroll
    for (int i = 0; i < 16; ++i) f[i] = feat[b * 16 + i];
    float nrm = 0.f;
    #pragma unroll
    for (int i = 0; i < 16; ++i) nrm += f[i] * f[i];
    float inv = 1.f / fmaxf(sqrtf(nrm), 1e-12f);
    float re[16], im[16];
    #pragma unroll
    for (int i = 0; i < 16; ++i) { re[i] = f[i] * inv; im[i] = 0.f; }

    #pragma unroll
    for (int layer = 0; layer < 2; ++layer) {
        #pragma unroll
        for (int q = 0; q < 4; ++q) {
            float phi = qw[(layer * 4 + q) * 3 + 0];
            float th  = qw[(layer * 4 + q) * 3 + 1];
            float om  = qw[(layer * 4 + q) * 3 + 2];
            float ct = cosf(0.5f * th), st = sinf(0.5f * th);
            float aa = 0.5f * (phi + om), bb = 0.5f * (phi - om);
            float ca = cosf(aa), sa = sinf(aa), cb = cosf(bb), sb = sinf(bb);
            // U00=ep*c, U01=-em*s, U10=conj(em)*s, U11=conj(ep)*c
            float U00r =  ca * ct, U00i = -sa * ct;
            float U01r = -cb * st, U01i = -sb * st;
            float U10r =  cb * st, U10i = -sb * st;
            float U11r =  ca * ct, U11i =  sa * ct;
            int mask = 1 << (3 - q);
            #pragma unroll
            for (int i = 0; i < 16; ++i) {
                if (i & mask) continue;
                int j = i | mask;
                float xr = re[i], xi = im[i], yr = re[j], yi = im[j];
                re[i] = U00r * xr - U00i * xi + U01r * yr - U01i * yi;
                im[i] = U00r * xi + U00i * xr + U01r * yi + U01i * yr;
                re[j] = U10r * xr - U10i * xi + U11r * yr - U11i * yi;
                im[j] = U10r * xi + U10i * xr + U11r * yi + U11i * yr;
            }
        }
        #pragma unroll
        for (int q = 0; q < 3; ++q) {                // CNOT q -> q+1
            int cm = 1 << (3 - q), tm = 1 << (2 - q);
            #pragma unroll
            for (int i = 0; i < 16; ++i) {
                if ((i & cm) && !(i & tm)) {
                    int j = i | tm;
                    float t = re[i]; re[i] = re[j]; re[j] = t;
                    t = im[i]; im[i] = im[j]; im[j] = t;
                }
            }
        }
    }
    float p[16];
    #pragma unroll
    for (int i = 0; i < 16; ++i) p[i] = re[i] * re[i] + im[i] * im[i];
    float qz[4];
    #pragma unroll
    for (int q = 0; q < 4; ++q) {
        float s = 0.f;
        #pragma unroll
        for (int i = 0; i < 16; ++i) s += (i & (1 << (3 - q))) ? -p[i] : p[i];
        qz[q] = s;
    }
    float z[10];
    #pragma unroll
    for (int o = 0; o < 10; ++o) {
        float s = fc4b[o];
        #pragma unroll
        for (int j = 0; j < 16; ++j) s += f[j] * fc4w[o * 20 + j];
        #pragma unroll
        for (int q = 0; q < 4; ++q) s += qz[q] * fc4w[o * 20 + 16 + q];
        z[o] = s;
    }
    float mx = z[0];
    #pragma unroll
    for (int o = 1; o < 10; ++o) mx = fmaxf(mx, z[o]);
    float se = 0.f;
    #pragma unroll
    for (int o = 0; o < 10; ++o) se += expf(z[o] - mx);
    float lse = logf(se);
    #pragma unroll
    for (int o = 0; o < 10; ++o) out[b * 10 + o] = z[o] - mx - lse;
}

// ---------- host ----------
extern "C" void kernel_launch(void* const* d_in, const int* in_sizes, int n_in,
                              void* d_out, int out_size, void* d_ws, size_t ws_size,
                              hipStream_t stream) {
    (void)in_sizes; (void)n_in; (void)out_size; (void)ws_size;
    const float* x    = (const float*)d_in[0];
    const float* c1w  = (const float*)d_in[1];
    const float* c1b  = (const float*)d_in[2];
    const float* c2w  = (const float*)d_in[3];
    const float* c2b  = (const float*)d_in[4];
    const float* fc1w = (const float*)d_in[5];
    const float* fc1b = (const float*)d_in[6];
    const float* fc2w = (const float*)d_in[7];
    const float* fc2b = (const float*)d_in[8];
    const float* fc3w = (const float*)d_in[9];
    const float* fc3b = (const float*)d_in[10];
    const float* qw   = (const float*)d_in[11];
    const float* fc4w = (const float*)d_in[12];
    const float* fc4b = (const float*)d_in[13];
    float* out = (float*)d_out;

    char* ws = (char*)d_ws;
    size_t off = 0;
    auto take = [&](size_t bytes) {
        char* p = ws + off;
        off = (off + bytes + 255) & ~(size_t)255;
        return p;
    };
    const size_t p1padBytes = 1024ULL * 16 * 18 * 32 * 2;           // padded NHWC bf16
    unsigned short* p1pad = (unsigned short*)take(p1padBytes);
    unsigned short* w2t   = (unsigned short*)take(9ULL * 64 * 32 * 2);
    unsigned short* p2b   = (unsigned short*)take(1024ULL * 3136 * 2);
    unsigned short* w1b   = (unsigned short*)take(128ULL * 3136 * 2);
    unsigned short* h1b   = (unsigned short*)take(1024ULL * 128 * 2);
    unsigned short* w2fb  = (unsigned short*)take(64ULL * 128 * 2);
    unsigned short* h2b   = (unsigned short*)take(1024ULL * 64 * 2);
    unsigned short* w3b   = (unsigned short*)take(16ULL * 64 * 2);
    float*          feat  = (float*)take(1024ULL * 16 * 4);

    hipMemsetAsync(p1pad, 0, p1padBytes, stream);   // zero halo (graph-capturable)
    pack_weights<<<1568, 256, 0, stream>>>(c2w, fc1w, fc2w, fc3w, w2t, w1b, w2fb, w3b);
    conv1_relu_pool<<<1024, 256, 0, stream>>>(x, c1w, c1b, p1pad);
    conv2_wmma_pool<<<1024 * 7, 256, 0, stream>>>(p1pad, w2t, c2b, p2b);
    gemm_bias_act<<<(64 * 8) / 4, 128, 0, stream>>>(p2b, w1b, fc1b, nullptr, h1b, 1024, 128, 3136, 1);
    gemm_bias_act<<<(64 * 4) / 4, 128, 0, stream>>>(h1b, w2fb, fc2b, nullptr, h2b, 1024, 64, 128, 1);
    gemm_bias_act<<<(64 * 1) / 4, 128, 0, stream>>>(h2b, w3b, fc3b, feat, nullptr, 1024, 16, 64, 1);
    quantum_fc4<<<4, 256, 0, stream>>>(feat, qw, fc4w, fc4b, out);
}
